// SwinBlock_13726715478133
// MI455X (gfx1250) — matrix operand order repaired and checked by reading, not compiled, not run on hardware
//
#include <hip/hip_runtime.h>

// ---------------------------------------------------------------------------
// Swin block (B=32, H=W=56, C=192, heads=6, d=32, WS=7 -> 2048 windows of 49
// tokens padded to 64) for gfx1250. All GEMMs run on v_wmma_f32_16x16x32_bf16
// (bf16 inputs, f32 accumulate); LN/softmax/GELU/residuals stay f32.
//
// GEMM structure (per block = 4 waves, one 64-row window / row-block):
//  * The 96x192 bf16 B panel for the current N-group is staged cooperatively
//    into LDS once per block (row stride padded 192->200 elements so the
//    16-lane fragment reads are LDS bank-conflict free), then all 4 waves
//    read B fragments with ds_load instead of 4x-redundant global loads.
//  * A fragments (6 x 8 = 48 VGPRs) are loaded once and reused across N-groups.
//  * Each wave accumulates 6 N-tiles (48 VGPRs) -> no scratch spills.
// ---------------------------------------------------------------------------

typedef __bf16 bf16;
typedef __attribute__((ext_vector_type(16))) __bf16 v16bf;
typedef __attribute__((ext_vector_type(8)))  __bf16 v8bf;
typedef __attribute__((ext_vector_type(8)))  float  v8f;

#define DEV __device__ __forceinline__

#define LDSB_LD 200  // padded LDS row stride (elements) for 192-col panels

// D = A(16x32 bf16) x B(32x16 bf16) + C, f32 accum.
DEV v8f wmma_bf16(v16bf a, v16bf b, v8f c) {
  return __builtin_amdgcn_wmma_f32_16x16x32_bf16(
      /*neg_a=*/false, a, /*neg_b=*/false, b,
      /*c_mod=*/(short)0, c, /*reuse_a=*/false, /*reuse_b=*/false);
}

// A-fragment (16x32, 16-bit): lane = m + 16*half; half 0 holds K {0..7,16..23},
// half 1 holds K {8..15,24..31} (ISA 7.12.2 16-bit A table). Two 16B loads.
DEV v16bf load_a(const bf16* p, int ld) {
  int lane = threadIdx.x & 31;
  const bf16* rp = p + (lane & 15) * ld + ((lane >> 4) << 3);
  v8bf lo = *(const v8bf*)(rp);
  v8bf hi = *(const v8bf*)(rp + 16);
  v16bf r;
#pragma unroll
  for (int i = 0; i < 8; ++i) { r[i] = lo[i]; r[i + 8] = hi[i]; }
  return r;
}

// B-fragment (32x16, 16-bit), source stored transposed (row = N index,
// columns = K): lane = n + 16*half; half 0 -> K {0..7,16..23}, half 1 -> K
// {8..15,24..31} (same K order as the A fragment). Two 16B loads.
DEV v16bf load_b(const bf16* p, int ld) {
  int lane = threadIdx.x & 31;
  const bf16* rp = p + (lane & 15) * ld + ((lane >> 4) << 3);
  v8bf lo = *(const v8bf*)(rp);
  v8bf hi = *(const v8bf*)(rp + 16);
  v16bf r;
#pragma unroll
  for (int i = 0; i < 8; ++i) { r[i] = lo[i]; r[i + 8] = hi[i]; }
  return r;
}

// Preload KS A-fragments (held in VGPRs across all N-groups).
template <int KS>
DEV void load_a_frags(const bf16* A0, int lda, v16bf (&a)[KS]) {
#pragma unroll
  for (int ks = 0; ks < KS; ++ks) a[ks] = load_a(A0 + ks * 32, lda);
}

// Cooperatively stage a 96x192 bf16 panel (row stride ldb) into LDS with a
// padded row stride of LDSB_LD. 128 threads x 18 iters x 16B.
DEV void stage_panel(const bf16* B0, int ldb, bf16* Bsh) {
  int tid = threadIdx.x;  // 0..127
#pragma unroll
  for (int it = 0; it < 18; ++it) {
    int idx = it * 128 + tid;        // vec8 index over 96*24
    int row = idx / 24, vc = idx % 24;
    *(v8bf*)(Bsh + row * LDSB_LD + vc * 8) =
        *(const v8bf*)(B0 + (size_t)row * ldb + vc * 8);
  }
}

// Group GEMM with register A-fragments and LDS-resident B panel.
template <int NT, int KS>
DEV void gemm_lds(const v16bf (&a)[KS], const bf16* Bsh, v8f (&acc)[NT]) {
#pragma unroll
  for (int ks = 0; ks < KS; ++ks)
#pragma unroll
    for (int t = 0; t < NT; ++t)
      acc[t] = wmma_bf16(a[ks], load_b(Bsh + t * 16 * LDSB_LD + ks * 32, LDSB_LD),
                         acc[t]);
}

// wave32 reductions
DEV float wave_sum(float v) {
#pragma unroll
  for (int m = 16; m >= 1; m >>= 1) v += __shfl_xor(v, m, 32);
  return v;
}
// reduce within 16-lane halves (C/D frag rows live in one half)
DEV float half_max(float v) {
#pragma unroll
  for (int m = 8; m >= 1; m >>= 1) v = fmaxf(v, __shfl_xor(v, m, 32));
  return v;
}
DEV float half_sum(float v) {
#pragma unroll
  for (int m = 8; m >= 1; m >>= 1) v += __shfl_xor(v, m, 32);
  return v;
}

// ---------------- weight f32 -> bf16 ----------------
__global__ void k_cvt(const float* __restrict__ s, bf16* __restrict__ d, int n) {
  int i = blockIdx.x * blockDim.x + threadIdx.x;
  if (i < n) d[i] = (bf16)s[i];
}

// ---------------- LN1 + window partition -> act[2048][64][192] bf16 --------
// one wave per padded row; rows 49..63 zero-filled.
__global__ void k_ln1(const float* __restrict__ x, const float* __restrict__ w,
                      const float* __restrict__ b, bf16* __restrict__ act) {
  int gw = blockIdx.x * (blockDim.x >> 5) + (threadIdx.x >> 5);
  int lane = threadIdx.x & 31;
  int win = gw >> 6, r = gw & 63;
  bf16* dst = act + (size_t)gw * 192;
  if (r >= 49) {
#pragma unroll
    for (int j = 0; j < 6; ++j) dst[lane + 32 * j] = (bf16)0.0f;
    return;
  }
  int bimg = win >> 6, wi = win & 63, wh = wi >> 3, ww = wi & 7;
  int l = (wh * 7 + r / 7) * 56 + (ww * 7 + r % 7);
  const float* src = x + ((size_t)bimg * 3136 + l) * 192;
  float v[6], s = 0.f;
#pragma unroll
  for (int j = 0; j < 6; ++j) { v[j] = src[lane + 32 * j]; s += v[j]; }
  float mu = wave_sum(s) * (1.f / 192.f);
  float q = 0.f;
#pragma unroll
  for (int j = 0; j < 6; ++j) { float d0 = v[j] - mu; q += d0 * d0; }
  float rs = rsqrtf(wave_sum(q) * (1.f / 192.f) + 1e-5f);
#pragma unroll
  for (int j = 0; j < 6; ++j) {
    int c = lane + 32 * j;
    dst[c] = (bf16)((v[j] - mu) * rs * w[c] + b[c]);
  }
}

// ---------------- QKV GEMM: act[w](64x192) x Wqkv^T(192x576) ----------------
// qkv layout per window (36864 el): q[h][tok64][32], k[h][tok64][32], v[h][32][tok64]
// Group g covers 96 cols: which = g>>1 (q/k/v), heads (g&1)*3 + t/2.
__global__ void k_qkv(const bf16* __restrict__ act, const bf16* __restrict__ Wq,
                      const float* __restrict__ bias, bf16* __restrict__ qkv) {
  __shared__ __attribute__((aligned(16))) bf16 Bsh[96 * LDSB_LD];
  int win = blockIdx.x;
  int wave = threadIdx.x >> 5, lane = threadIdx.x & 31;
  const bf16* A0 = act + (size_t)win * 64 * 192 + wave * 16 * 192;
  int n = lane & 15, half = lane >> 4;
  const v8f vzero = {};
  v16bf a[6];
  load_a_frags<6>(A0, 192, a);  // A reused across all 6 groups
  for (int g = 0; g < 6; ++g) {
    __syncthreads();  // previous group's readers done before restage
    stage_panel(Wq + (size_t)(g * 96) * 192, 192, Bsh);
    __syncthreads();
    v8f acc[6];
#pragma unroll
    for (int t = 0; t < 6; ++t) acc[t] = vzero;
    gemm_lds<6, 6>(a, Bsh, acc);
    int which = g >> 1, sub = g & 1;  // wave-uniform
    float scale = (which == 0) ? 0.17677669529663689f : 1.0f;  // 1/sqrt(32)
    bf16* base = qkv + (size_t)win * 36864 + which * 12288;
#pragma unroll
    for (int t = 0; t < 6; ++t) {
      int head = sub * 3 + (t >> 1);     // tile spans 16 cols; head uniform
      int dcol = ((t & 1) << 4) + n;     // 0..31
      float bj = bias[g * 96 + t * 16 + n];
      bf16* hb = base + head * 2048;
#pragma unroll
      for (int rr = 0; rr < 8; ++rr) {
        int row = wave * 16 + rr + 8 * half;
        float val = (acc[t][rr] + bj) * scale;
        if (which < 2) hb[row * 32 + dcol] = (bf16)val;   // q,k: [tok][d]
        else           hb[dcol * 64 + row] = (bf16)val;   // v  : [d][tok]
      }
    }
  }
}

// ---------------- attention: per (window, head) -----------------------------
__global__ void k_attn(const bf16* __restrict__ qkv, bf16* __restrict__ Oact) {
  __shared__ __attribute__((aligned(16))) bf16 Psh[64 * 64];
  int win = blockIdx.x / 6, head = blockIdx.x % 6;
  int wave = threadIdx.x >> 5, lane = threadIdx.x & 31;
  int n = lane & 15, half = lane >> 4;
  const bf16* q  = qkv + (size_t)win * 36864 + head * 2048;           // [64][32]
  const bf16* kk = qkv + (size_t)win * 36864 + 12288 + head * 2048;   // [64][32]
  const bf16* vT = qkv + (size_t)win * 36864 + 24576 + head * 2048;   // [32][64]

  // S = q k^T : wave owns M-tile `wave` (16 rows x 64 cols), K=32 = 1 step
  v16bf aq = load_a(q + wave * 16 * 32, 32);
  v8f s[4];
#pragma unroll
  for (int nt = 0; nt < 4; ++nt) {
    v8f z = {};
    s[nt] = wmma_bf16(aq, load_b(kk + nt * 16 * 32, 32), z);
  }
  // mask padded key columns (col = 48+n valid only for n==0 in tile 3)
  if (n >= 1) {
#pragma unroll
    for (int rr = 0; rr < 8; ++rr) s[3][rr] = -1e30f;
  }
  // softmax per row (row values live in one 16-lane half across 4 tiles)
#pragma unroll
  for (int rr = 0; rr < 8; ++rr) {
    float mx = fmaxf(fmaxf(s[0][rr], s[1][rr]), fmaxf(s[2][rr], s[3][rr]));
    mx = half_max(mx);
    float e[4], sm = 0.f;
#pragma unroll
    for (int nt = 0; nt < 4; ++nt) { e[nt] = expf(s[nt][rr] - mx); sm += e[nt]; }
    sm = half_sum(sm);
    float inv = 1.0f / sm;
    int row = wave * 16 + rr + 8 * half;
#pragma unroll
    for (int nt = 0; nt < 4; ++nt) Psh[row * 64 + nt * 16 + n] = (bf16)(e[nt] * inv);
  }
  __syncthreads();  // re-layout C/D-frag -> A-frag through LDS

  // O = P(64x64) @ V(64x32): 2 N-tiles x 2 K-steps
  v8f acc[2];
  {
    const v8f vzero = {};
    acc[0] = vzero; acc[1] = vzero;
  }
#pragma unroll
  for (int ks = 0; ks < 2; ++ks) {
    v16bf pa = load_a(Psh + wave * 16 * 64 + ks * 32, 64);
#pragma unroll
    for (int nt = 0; nt < 2; ++nt)
      acc[nt] = wmma_bf16(pa, load_b(vT + nt * 16 * 64 + ks * 32, 64), acc[nt]);
  }
#pragma unroll
  for (int nt = 0; nt < 2; ++nt)
#pragma unroll
    for (int rr = 0; rr < 8; ++rr) {
      int row = wave * 16 + rr + 8 * half;
      Oact[((size_t)win * 64 + row) * 192 + head * 32 + nt * 16 + n] = (bf16)acc[nt][rr];
    }
}

// ---------------- out-proj + residual -> attnx (f32, token-major) -----------
__global__ void k_outproj(const bf16* __restrict__ Oact, const bf16* __restrict__ Wo,
                          const float* __restrict__ bias, const float* __restrict__ x,
                          float* __restrict__ attnx) {
  __shared__ __attribute__((aligned(16))) bf16 Bsh[96 * LDSB_LD];
  int win = blockIdx.x;
  int wave = threadIdx.x >> 5, lane = threadIdx.x & 31;
  int n = lane & 15, half = lane >> 4;
  int bimg = win >> 6, wi = win & 63, wh = wi >> 3, ww = wi & 7;
  const bf16* A0 = Oact + (size_t)win * 64 * 192 + wave * 16 * 192;
  const v8f vzero = {};
  v16bf a[6];
  load_a_frags<6>(A0, 192, a);
  for (int g = 0; g < 2; ++g) {  // 2 groups x 6 tiles = 192 cols
    __syncthreads();
    stage_panel(Wo + (size_t)(g * 96) * 192, 192, Bsh);
    __syncthreads();
    v8f acc[6];
#pragma unroll
    for (int t = 0; t < 6; ++t) acc[t] = vzero;
    gemm_lds<6, 6>(a, Bsh, acc);
#pragma unroll
    for (int t = 0; t < 6; ++t) {
      int col = g * 96 + t * 16 + n;
      float bj = bias[col];
#pragma unroll
      for (int rr = 0; rr < 8; ++rr) {
        int r = wave * 16 + rr + 8 * half;
        if (r < 49) {  // window reverse: drop pad rows
          int l = (wh * 7 + r / 7) * 56 + (ww * 7 + r % 7);
          size_t idx = ((size_t)bimg * 3136 + l) * 192 + col;
          attnx[idx] = x[idx] + acc[t][rr] + bj;
        }
      }
    }
  }
}

// ---------------- LN2 -> h2 bf16 (token-major, 100352 x 192) ----------------
__global__ void k_ln2(const float* __restrict__ ax, const float* __restrict__ w,
                      const float* __restrict__ b, bf16* __restrict__ h2) {
  int gw = blockIdx.x * (blockDim.x >> 5) + (threadIdx.x >> 5);
  int lane = threadIdx.x & 31;
  const float* src = ax + (size_t)gw * 192;
  bf16* dst = h2 + (size_t)gw * 192;
  float v[6], s = 0.f;
#pragma unroll
  for (int j = 0; j < 6; ++j) { v[j] = src[lane + 32 * j]; s += v[j]; }
  float mu = wave_sum(s) * (1.f / 192.f);
  float q = 0.f;
#pragma unroll
  for (int j = 0; j < 6; ++j) { float d0 = v[j] - mu; q += d0 * d0; }
  float rs = rsqrtf(wave_sum(q) * (1.f / 192.f) + 1e-5f);
#pragma unroll
  for (int j = 0; j < 6; ++j) {
    int c = lane + 32 * j;
    dst[c] = (bf16)((v[j] - mu) * rs * w[c] + b[c]);
  }
}

// ---------------- MLP1: h2(100352x192) x W1^T -> GELU -> hid bf16 -----------
__global__ void k_mlp1(const bf16* __restrict__ h2, const bf16* __restrict__ W1,
                       const float* __restrict__ b1, bf16* __restrict__ hid) {
  __shared__ __attribute__((aligned(16))) bf16 Bsh[96 * LDSB_LD];
  size_t row0 = (size_t)blockIdx.x * 64 + (size_t)(threadIdx.x >> 5) * 16;
  int lane = threadIdx.x & 31, n = lane & 15, half = lane >> 4;
  const bf16* A0 = h2 + row0 * 192;
  const v8f vzero = {};
  v16bf a[6];
  load_a_frags<6>(A0, 192, a);  // A reused across all 8 groups
  for (int g = 0; g < 8; ++g) {  // 8 groups x 6 tiles = 768 cols
    __syncthreads();
    stage_panel(W1 + (size_t)(g * 96) * 192, 192, Bsh);
    __syncthreads();
    v8f acc[6];
#pragma unroll
    for (int t = 0; t < 6; ++t) acc[t] = vzero;
    gemm_lds<6, 6>(a, Bsh, acc);
#pragma unroll
    for (int t = 0; t < 6; ++t) {
      int col = g * 96 + t * 16 + n;
      float bj = b1[col];
#pragma unroll
      for (int rr = 0; rr < 8; ++rr) {
        float u = acc[t][rr] + bj;
        float gl = 0.5f * u * (1.0f + erff(u * 0.70710678118654752f));  // exact GELU
        hid[(row0 + rr + 8 * half) * 768 + col] = (bf16)gl;
      }
    }
  }
}

// ---------------- MLP2 + residual -> out f32 --------------------------------
__global__ void k_mlp2(const bf16* __restrict__ hid, const bf16* __restrict__ W2,
                       const float* __restrict__ b2, const float* __restrict__ ax,
                       float* __restrict__ out) {
  __shared__ __attribute__((aligned(16))) bf16 Bsh[96 * LDSB_LD];
  size_t row0 = (size_t)blockIdx.x * 64 + (size_t)(threadIdx.x >> 5) * 16;
  int lane = threadIdx.x & 31, n = lane & 15, half = lane >> 4;
  const bf16* A0 = hid + row0 * 768;
  const v8f vzero = {};
  for (int g = 0; g < 2; ++g) {    // 2 N-groups x 6 tiles = 192 cols
    v8f acc[6];
#pragma unroll
    for (int t = 0; t < 6; ++t) acc[t] = vzero;
    for (int kc = 0; kc < 4; ++kc) {  // K = 768 in 4 chunks of 6 K-steps
      __syncthreads();
      stage_panel(W2 + (size_t)(g * 96) * 768 + kc * 192, 768, Bsh);
      __syncthreads();
      v16bf a[6];
      load_a_frags<6>(A0 + kc * 192, 768, a);
      gemm_lds<6, 6>(a, Bsh, acc);
    }
#pragma unroll
    for (int t = 0; t < 6; ++t) {
      int col = g * 96 + t * 16 + n;
      float bj = b2[col];
#pragma unroll
      for (int rr = 0; rr < 8; ++rr) {
        size_t idx = (row0 + rr + 8 * half) * 192 + col;
        out[idx] = ax[idx] + acc[t][rr] + bj;
      }
    }
  }
}

// ---------------------------------------------------------------------------
extern "C" void kernel_launch(void* const* d_in, const int* in_sizes, int n_in,
                              void* d_out, int out_size, void* d_ws, size_t ws_size,
                              hipStream_t stream) {
  (void)in_sizes; (void)n_in; (void)out_size; (void)ws_size;
  const float* x    = (const float*)d_in[0];
  const float* n1w  = (const float*)d_in[1];
  const float* n1b  = (const float*)d_in[2];
  const float* Wqkv = (const float*)d_in[3];
  const float* bqkv = (const float*)d_in[4];
  const float* Wout = (const float*)d_in[5];
  const float* bout = (const float*)d_in[6];
  const float* n2w  = (const float*)d_in[7];
  const float* n2b  = (const float*)d_in[8];
  const float* W1   = (const float*)d_in[9];
  const float* b1   = (const float*)d_in[10];
  const float* W2   = (const float*)d_in[11];
  const float* b2   = (const float*)d_in[12];
  float* out = (float*)d_out;

  // workspace layout (phases reuse regions; peak ~282.5 MB):
  //  buf0 : qkv [2048][3][6][...] bf16 (151MB), later MLP hidden (154.1MB)
  //  act  : LN1 windows -> attn output -> LN2 output (50.3MB)
  //  attnx: x + attention path, f32 (77.1MB)
  //  wb   : bf16 weights (0.9MB)
  char* ws = (char*)d_ws;
  bf16*  buf0  = (bf16*)(ws);
  bf16*  act   = (bf16*)(ws + 154140672u);
  float* attnx = (float*)(ws + 154140672u + 50331648u);
  bf16*  wb    = (bf16*)(ws + 154140672u + 50331648u + 77070336u);
  bf16* wq = wb;            // 576*192
  bf16* wo = wb + 110592;   // 192*192
  bf16* w1 = wb + 147456;   // 768*192
  bf16* w2 = wb + 294912;   // 192*768

  k_cvt<<<(110592 + 255) / 256, 256, 0, stream>>>(Wqkv, wq, 110592);
  k_cvt<<<(36864  + 255) / 256, 256, 0, stream>>>(Wout, wo, 36864);
  k_cvt<<<(147456 + 255) / 256, 256, 0, stream>>>(W1,   w1, 147456);
  k_cvt<<<(147456 + 255) / 256, 256, 0, stream>>>(W2,   w2, 147456);

  k_ln1<<<16384, 256, 0, stream>>>(x, n1w, n1b, act);          // 2048*64 rows
  k_qkv<<<2048, 128, 0, stream>>>(act, wq, bqkv, buf0);
  k_attn<<<2048 * 6, 128, 0, stream>>>(buf0, act);             // overwrites act
  k_outproj<<<2048, 128, 0, stream>>>(act, wo, bout, x, attnx);
  k_ln2<<<12544, 256, 0, stream>>>(attnx, n2w, n2b, act);      // h2 in act
  k_mlp1<<<1568, 128, 0, stream>>>(act, w1, b1, buf0);         // hidden in buf0
  k_mlp2<<<1568, 128, 0, stream>>>(buf0, w2, b2, attnx, out);
}
